// CausalSelfAttention_15668040696055
// MI455X (gfx1250) — matrix operand order repaired and checked by reading, not compiled, not run on hardware
//
#include <hip/hip_runtime.h>
#include <hip/hip_bf16.h>

typedef _Float16 half_t;
typedef __attribute__((ext_vector_type(8)))  _Float16 v8h;
typedef __attribute__((ext_vector_type(16))) _Float16 v16h;
typedef __attribute__((ext_vector_type(8)))  float    v8f;

#define DIM    1024
#define NHEAD  16
#define HDIM   64
#define SEQ    4096
#define ASCALE 0.12f
#define RMSEPS 1.1920929e-07f

// Build a 16-element f16 fragment from two contiguous 8-element (16B) runs.
static __device__ __forceinline__ v16h frag16(const half_t* p, int o0, int o1) {
    v8h lo = *(const v8h*)(p + o0);
    v8h hi = *(const v8h*)(p + o1);
    return __builtin_shufflevector(lo, hi, 0,1,2,3,4,5,6,7,8,9,10,11,12,13,14,15);
}

static __device__ __forceinline__ v8f wmma_f16(v16h a, v16h b, v8f c) {
    // D = A(16x32) * B(32x16) + C, f32 accumulate
    return __builtin_amdgcn_wmma_f32_16x16x32_f16(false, a, false, b, (short)0, c,
                                                  false, false);
}

// ---------------------------------------------------------------------------
// GEMM: C[M,N] = A[M,K] * B[N,K]^T   (A: TA (f32 or f16), B: f32, C: f32)
// Block tile 128x128, 256 threads = 8 waves; wave (rc,cc) computes a 32x64
// sub-tile => 8 WMMA per K-step per wave against 6 fragment loads.
// ---------------------------------------------------------------------------
template <typename TA>
__global__ __launch_bounds__(256)
void gemm_abt(const TA* __restrict__ A, const float* __restrict__ B,
              float* __restrict__ C, int M, int N, int K)
{
    __shared__ half_t As[128 * 32];   // [m][k] row-major           (8 KB)
    __shared__ half_t Bs[32 * 128];   // [n][k] row-major           (8 KB)

    const int tid  = threadIdx.x;
    const int lane = tid & 31;
    const int wave = tid >> 5;
    const int ln   = lane & 15;
    const int hf   = lane >> 4;
    const int rc   = wave & 3;        // row chunk: 32 rows
    const int cc   = wave >> 2;       // col chunk: 64 cols
    const int m0   = blockIdx.y * 128;
    const int n0   = blockIdx.x * 128;

    const int lrow = tid >> 1;        // 0..127
    const int lk   = (tid & 1) * 16;  // 0 or 16

    v8f acc[2][4] = {};

    for (int ks = 0; ks < K; ks += 32) {
        { // stage A tile (128x32), convert to f16
            const TA* src = A + (size_t)(m0 + lrow) * K + ks + lk;
            if (ks + 32 < K) __builtin_prefetch(src + 32, 0, 1);
            half_t* dst = As + lrow * 32 + lk;
            #pragma unroll
            for (int i = 0; i < 16; ++i) dst[i] = (half_t)src[i];
        }
        { // stage B tile: Bs[n][k] = B[(n0+n)*K + ks + k]
            const float* src = B + (size_t)(n0 + lrow) * K + ks + lk;
            if (ks + 32 < K) __builtin_prefetch(src + 32, 0, 1);
            #pragma unroll
            for (int i = 0; i < 16; ++i) Bs[lrow * 32 + lk + i] = (half_t)src[i];
        }
        __syncthreads();

        // B fragments: lane = n column, split K runs
        v16h bf[4];
        #pragma unroll
        for (int ct = 0; ct < 4; ++ct)
            bf[ct] = frag16(Bs + (cc * 64 + ct * 16 + ln) * 32, hf * 8, hf * 8 + 16);

        const int koff = hf * 8;
        #pragma unroll
        for (int rt = 0; rt < 2; ++rt) {
            // A fragment: lane = m row, split K runs
            v16h af = frag16(As + (rc * 32 + rt * 16 + ln) * 32, koff, koff + 16);
            #pragma unroll
            for (int ct = 0; ct < 4; ++ct)
                acc[rt][ct] = wmma_f16(af, bf[ct], acc[rt][ct]);
        }
        __syncthreads();
    }

    #pragma unroll
    for (int rt = 0; rt < 2; ++rt) {
        #pragma unroll
        for (int ct = 0; ct < 4; ++ct) {
            const int cn = n0 + cc * 64 + ct * 16 + ln;
            #pragma unroll
            for (int v = 0; v < 8; ++v) {
                int cm = m0 + rc * 32 + rt * 16 + v + 8 * hf;  // C layout
                C[(size_t)cm * N + cn] = acc[rt][ct][v];
            }
        }
    }
}

// ---------------------------------------------------------------------------
// Prep: split qkv, blend V with ve, RMS-norm + RoPE on Q/K, downcast to f16.
// Q: [h][t][d] row-major, K: [h][t][d] row-major, V: [h][d][t] transposed.
// One thread per (t, h).
// ---------------------------------------------------------------------------
__global__ __launch_bounds__(256)
void prep_kernel(const float* __restrict__ qkv, const float* __restrict__ ve,
                 const float* __restrict__ lambdas,
                 half_t* __restrict__ Qh, half_t* __restrict__ Kt,
                 half_t* __restrict__ Vh)
{
    const int idx = blockIdx.x * 256 + threadIdx.x;   // SEQ*NHEAD
    const int t = idx >> 4;
    const int h = idx & 15;
    const float* row = qkv + (size_t)t * (3 * DIM) + h * HDIM;
    const float l0 = lambdas[0], l1 = lambdas[1];

    { // V blend
        const float* vsrc  = row + 2 * DIM;
        const float* vesrc = ve + (size_t)t * DIM + h * HDIM;
        half_t* vd = Vh + (size_t)h * HDIM * SEQ + t;
        #pragma unroll
        for (int d = 0; d < HDIM; ++d)
            vd[(size_t)d * SEQ] = (half_t)(l0 * vsrc[d] + l1 * vesrc[d]);
    }

    // RoPE tables: af[i] = (1/1024)^(i/15) for i<16 else 0
    float cs[32], sn[32];
    #pragma unroll
    for (int d = 0; d < 32; ++d) {
        float af = (d < 16) ? exp2f(-10.0f * (float)d * (1.0f / 15.0f)) : 0.0f;
        float th = (float)t * af;
        sincosf(th, &sn[d], &cs[d]);
    }

    { // Q: rms-norm + rope, store row-major
        float x[HDIM]; float ms = 0.0f;
        #pragma unroll
        for (int d = 0; d < HDIM; ++d) { x[d] = row[d]; ms += x[d] * x[d]; }
        float r = rsqrtf(ms * (1.0f / HDIM) + RMSEPS);
        half_t* qd = Qh + ((size_t)h * SEQ + t) * HDIM;
        #pragma unroll
        for (int d = 0; d < 32; ++d) {
            float x1 = x[d] * r, x2 = x[d + 32] * r;
            qd[d]      = (half_t)( x1 * cs[d] + x2 * sn[d]);
            qd[d + 32] = (half_t)(-x1 * sn[d] + x2 * cs[d]);
        }
    }
    { // K: rms-norm + rope, store row-major [h][t][d]
        const float* src = row + DIM;
        float x[HDIM]; float ms = 0.0f;
        #pragma unroll
        for (int d = 0; d < HDIM; ++d) { x[d] = src[d]; ms += x[d] * x[d]; }
        float r = rsqrtf(ms * (1.0f / HDIM) + RMSEPS);
        #pragma unroll
        for (int d = 0; d < 32; ++d) {
            float x1 = x[d] * r, x2 = x[d + 32] * r;
            Kt[((size_t)h * SEQ + t) * HDIM + d]      = (half_t)( x1 * cs[d] + x2 * sn[d]);
            Kt[((size_t)h * SEQ + t) * HDIM + d + 32] = (half_t)(-x1 * sn[d] + x2 * cs[d]);
        }
    }
}

// ---------------------------------------------------------------------------
// Flash attention: grid (SEQ/64, NHEAD), 128 threads = 4 waves.
// Wave w: 16 q rows [q0+16w, q0+16w+16), full 64-dim output, online softmax
// over 32-key blocks. Scores/PV via f16 WMMA, f32 accumulate.
// ---------------------------------------------------------------------------
__global__ __launch_bounds__(128)
void attn_kernel(const half_t* __restrict__ Qh, const half_t* __restrict__ Kt,
                 const half_t* __restrict__ Vh, half_t* __restrict__ Y)
{
    __shared__ half_t plds[4 * 16 * 32];   // per-wave P tile (16 q x 32 keys)

    const int tid  = threadIdx.x;
    const int lane = tid & 31;
    const int wave = tid >> 5;
    const int ln   = lane & 15;
    const int hf   = lane >> 4;
    const int q0   = blockIdx.x * 64;
    const int h    = blockIdx.y;
    const int qrow = q0 + wave * 16;

    // Q fragments for the two K-dim steps (A layout)
    const half_t* qbase = Qh + ((size_t)h * SEQ + qrow + ln) * HDIM;
    const v16h qf0 = frag16(qbase, hf * 8,      hf * 8 + 16);
    const v16h qf1 = frag16(qbase, hf * 8 + 32, hf * 8 + 48);

    v8f o[4] = {};
    float m[8], l[8];
    #pragma unroll
    for (int v = 0; v < 8; ++v) { m[v] = -1e30f; l[v] = 0.0f; }

    half_t* pb = plds + wave * 512;
    const int nblk = q0 / 32 + 2;   // causal: keys 0 .. q0+63

    for (int j = 0; j < nblk; ++j) {
        // S(16x32) = Q(16x64) * K^T(64x32): 2 key-subtiles x 2 K-steps
        v8f s0 = {}, s1 = {};
        const half_t* krow0 = Kt + ((size_t)h * SEQ + j * 32 + ln) * HDIM;
        const half_t* krow1 = krow0 + 16 * HDIM;
        s0 = wmma_f16(qf0, frag16(krow0, hf * 8,      hf * 8 + 16), s0);
        s1 = wmma_f16(qf0, frag16(krow1, hf * 8,      hf * 8 + 16), s1);
        s0 = wmma_f16(qf1, frag16(krow0, hf * 8 + 32, hf * 8 + 48), s0);
        s1 = wmma_f16(qf1, frag16(krow1, hf * 8 + 32, hf * 8 + 48), s1);

        // online softmax (rows live in 16-lane halves; reduce with shfl_xor)
        float alpha[8];
        #pragma unroll
        for (int v = 0; v < 8; ++v) {
            const int qr  = qrow + v + 8 * hf;
            const int k0i = j * 32 + ln;
            const bool ok0 = (k0i <= qr), ok1 = (k0i + 16 <= qr);
            float sv0 = s0[v] * ASCALE;
            float sv1 = s1[v] * ASCALE;
            float mx = fmaxf(ok0 ? sv0 : -1e30f, ok1 ? sv1 : -1e30f);
            #pragma unroll
            for (int d = 1; d < 16; d <<= 1) mx = fmaxf(mx, __shfl_xor(mx, d, 32));
            float mn = fmaxf(m[v], mx);
            alpha[v] = __expf(m[v] - mn);
            float p0 = ok0 ? __expf(sv0 - mn) : 0.0f;
            float p1 = ok1 ? __expf(sv1 - mn) : 0.0f;
            float rs = p0 + p1;
            #pragma unroll
            for (int d = 1; d < 16; d <<= 1) rs += __shfl_xor(rs, d, 32);
            l[v] = l[v] * alpha[v] + rs;
            m[v] = mn;
            s0[v] = p0; s1[v] = p1;   // reuse as P
        }
        #pragma unroll
        for (int dt = 0; dt < 4; ++dt)
            #pragma unroll
            for (int v = 0; v < 8; ++v) o[dt][v] *= alpha[v];

        // P (C layout, f32) -> LDS as f16 row-major -> reload in A layout
        #pragma unroll
        for (int v = 0; v < 8; ++v) {
            pb[(v + 8 * hf) * 32 + ln]      = (half_t)s0[v];
            pb[(v + 8 * hf) * 32 + 16 + ln] = (half_t)s1[v];
        }
        __syncthreads();
        v16h pf = frag16(pb + ln * 32, hf * 8, hf * 8 + 16);

        // O(16x64) += P(16x32) * V(32x64)
        #pragma unroll
        for (int dt = 0; dt < 4; ++dt) {
            const half_t* vb = Vh + ((size_t)h * HDIM + dt * 16 + ln) * SEQ + j * 32;
            o[dt] = wmma_f16(pf, frag16(vb, hf * 8, hf * 8 + 16), o[dt]);
        }
        __syncthreads();
    }

    // normalize and store Y (f16, row-major [t][DIM])
    #pragma unroll
    for (int v = 0; v < 8; ++v) {
        float inv = 1.0f / l[v];
        int t = qrow + v + 8 * hf;
        #pragma unroll
        for (int dt = 0; dt < 4; ++dt)
            Y[(size_t)t * DIM + h * HDIM + dt * 16 + ln] = (half_t)(o[dt][v] * inv);
    }
}

// ---------------------------------------------------------------------------
extern "C" void kernel_launch(void* const* d_in, const int* in_sizes, int n_in,
                              void* d_out, int out_size, void* d_ws, size_t ws_size,
                              hipStream_t stream)
{
    const float* x    = (const float*)d_in[0];
    const float* ve   = (const float*)d_in[1];
    const float* qkvw = (const float*)d_in[2];   // [3*DIM, DIM]
    const float* lam  = (const float*)d_in[3];   // [2]
    const float* cpw  = (const float*)d_in[4];   // [DIM, DIM]
    float* out = (float*)d_out;

    char*   ws  = (char*)d_ws;
    float*  qkv = (float*)ws;                                   // 48 MB f32
    half_t* Qh  = (half_t*)(ws + (size_t)SEQ * 3 * DIM * 4);    // 8 MB
    half_t* Kt  = Qh + (size_t)NHEAD * SEQ * HDIM;              // 8 MB
    half_t* Vh  = Kt + (size_t)NHEAD * SEQ * HDIM;              // 8 MB
    half_t* Yh  = Vh + (size_t)NHEAD * SEQ * HDIM;              // 8 MB

    // 1) qkv = x @ qkv_w^T
    gemm_abt<float><<<dim3(3 * DIM / 128, SEQ / 128), 256, 0, stream>>>(
        x, qkvw, qkv, SEQ, 3 * DIM, DIM);
    // 2) split / blend / norm / rope
    prep_kernel<<<dim3(SEQ * NHEAD / 256), 256, 0, stream>>>(
        qkv, ve, lam, Qh, Kt, Vh);
    // 3) flash attention
    attn_kernel<<<dim3(SEQ / 64, NHEAD), 128, 0, stream>>>(Qh, Kt, Vh, Yh);
    // 4) out = y @ c_proj^T
    gemm_abt<half_t><<<dim3(DIM / 128, SEQ / 128), 256, 0, stream>>>(
        Yh, cpw, out, SEQ, DIM, DIM);
}
